// VitEncoder_6296422056328
// MI455X (gfx1250) — matrix operand order repaired and checked by reading, not compiled, not run on hardware
//
#include <hip/hip_runtime.h>
#include <math.h>

#define BB   32
#define SS   197
#define DDIM 384
#define HH   6
#define HDD  64
#define HID  1024
#define SD   (SS*DDIM)          // 75648
#define NTOK (BB*SS*DDIM)       // 2420736 == B*H*S*HD
#define EPSF 1e-5f
#define SPAD 224                // S padded to multiple of 32
#define MT   13                 // ceil(197/16)
#define KC1  (SD/32)            // 2364 K-chunks in MLP1
#define KSPLIT 48
#define KPER ((KC1 + KSPLIT - 1)/KSPLIT)   // 50

typedef __attribute__((ext_vector_type(16))) __bf16 v16bf;
typedef __attribute__((ext_vector_type(8)))  float  v8f;
typedef unsigned int u32x4 __attribute__((ext_vector_type(4)));
typedef int          i32x8 __attribute__((ext_vector_type(8)));
typedef int          i32x4 __attribute__((ext_vector_type(4)));

static __device__ inline v8f wmma_bf16(v16bf a, v16bf b, v8f c) {
  return __builtin_amdgcn_wmma_f32_16x16x32_bf16(false, a, false, b, (short)0, c, false, false);
}

// A-matrix 16x32 bf16; row index clamped (C rows >= mvalid are don't-care, never stored)
static __device__ inline v16bf load_a_bf(const __bf16* src, size_t lda, int mvalid) {
  int lane = threadIdx.x & 31;
  int m = lane & 15;
  m = (m < mvalid) ? m : (mvalid - 1);
  int kb = (lane >> 4) << 4;
  const __bf16* p = src + (size_t)m * lda + kb;
  v16bf a;
#pragma unroll
  for (int j = 0; j < 8; ++j) a[j] = p[j];
#pragma unroll
  for (int j = 0; j < 8; ++j) a[8 + j] = p[8 + j];
  return a;
}

static __device__ inline v16bf load_a_full(const __bf16* src, size_t lda) {
  int lane = threadIdx.x & 31;
  int m = lane & 15;
  int kb = (lane >> 4) << 4;
  const __bf16* p = src + (size_t)m * lda + kb;
  v16bf a;
#pragma unroll
  for (int j = 0; j < 8; ++j) a[j] = p[j];
#pragma unroll
  for (int j = 0; j < 8; ++j) a[8 + j] = p[8 + j];
  return a;
}

// B-matrix 32x16 (KxN) bf16, strided walk (used only where layout forces it)
static __device__ inline v16bf load_b_bf(const __bf16* src, size_t ldb) {
  int lane = threadIdx.x & 31;
  int n = lane & 15;
  int kb = (lane >> 4) << 4;
  v16bf b;
#pragma unroll
  for (int j = 0; j < 16; ++j) b[j] = src[(size_t)(kb + j) * ldb + n];
  return b;
}

// B-matrix with clamped K (clamp keeps values finite; matching A columns are zero)
static __device__ inline v16bf load_b_clamp(const __bf16* src, size_t ldb, int kvalid) {
  int lane = threadIdx.x & 31;
  int n = lane & 15;
  int kb = (lane >> 4) << 4;
  v16bf b;
#pragma unroll
  for (int j = 0; j < 16; ++j) {
    int k = kb + j;
    k = (k < kvalid) ? k : (kvalid - 1);
    b[j] = src[(size_t)k * ldb + n];
  }
  return b;
}

// B[k][n] = src[n][k]  (contiguous 32B per lane -> wide loads)
static __device__ inline v16bf load_bT_bf(const __bf16* src, size_t ld) {
  int lane = threadIdx.x & 31;
  int n = lane & 15;
  int kb = (lane >> 4) << 4;
  v16bf b;
  const __bf16* p = src + (size_t)n * ld + kb;
#pragma unroll
  for (int j = 0; j < 16; ++j) b[j] = p[j];
  return b;
}

// B-matrix from fp32 memory, converted in flight
static __device__ inline v16bf load_b_f32(const float* src, size_t ldb) {
  int lane = threadIdx.x & 31;
  int n = lane & 15;
  int kb = (lane >> 4) << 4;
  v16bf b;
#pragma unroll
  for (int j = 0; j < 16; ++j) b[j] = (__bf16)src[(size_t)(kb + j) * ldb + n];
  return b;
}

static __device__ inline float gelu_exact(float v) {
  return 0.5f * v * (1.0f + erff(v * 0.70710678118654752f));
}

// ---- Tensor Data Mover: 1-D tile (nelem bf16 elements) global -> LDS ----
// clang-23 toolchain: 6-arg form (g0 u32x4, g1 i32x8, g2 i32x4, g3 i32x4, g4 i32x8, cpol)
static __device__ inline void tdm_load_1d_bf16(const __bf16* gsrc, unsigned lds_byte_addr,
                                               int nelem) {
  unsigned long long ga = (unsigned long long)(size_t)gsrc;
  u32x4 g0;
  g0[0] = 1u;                                        // count=1, user mode
  g0[1] = lds_byte_addr;                             // lds_addr
  g0[2] = (unsigned)(ga & 0xFFFFFFFFu);              // global_addr[31:0]
  g0[3] = (unsigned)((ga >> 32) & 0x1FFFFFFu) | (2u << 30);  // global_addr[56:32], type=2
  i32x8 g1;
  unsigned ne = (unsigned)nelem;
  g1[0] = (int)(1u << 16);                           // data_size=1 (2 bytes), mask=0
  g1[1] = (int)((ne & 0xFFFFu) << 16);               // tensor_dim0[15:0] @ bits 63:48
  g1[2] = (int)(((ne >> 16) & 0xFFFFu) | (1u << 16));// tensor_dim0[31:16], tensor_dim1=1 (low)
  g1[3] = (int)((ne & 0xFFFFu) << 16);               // tensor_dim1 hi=0, tile_dim0=nelem
  g1[4] = 0;                                         // tile_dim1=0, tile_dim2=0
  g1[5] = (int)ne;                                   // tensor_dim0_stride low
  g1[6] = 0;
  g1[7] = 0;
  i32x4 gz4 = {0, 0, 0, 0};
  i32x8 gz8 = {0, 0, 0, 0, 0, 0, 0, 0};
  __builtin_amdgcn_tensor_load_to_lds(g0, g1, gz4, gz4, gz8, 0);
#if defined(__has_builtin)
#if __has_builtin(__builtin_amdgcn_s_wait_tensorcnt)
  __builtin_amdgcn_s_wait_tensorcnt(0);
#else
  asm volatile("s_wait_tensorcnt 0x0" ::: "memory");
#endif
#else
  asm volatile("s_wait_tensorcnt 0x0" ::: "memory");
#endif
}

// ---------------- kernels ----------------

__global__ void k_zero(float* p, int n) {
  int i = blockIdx.x * blockDim.x + threadIdx.x;
  int st = gridDim.x * blockDim.x;
  for (; i < n; i += st) p[i] = 0.f;
}

// QKV weights -> bf16, transposed to [proj][h][e][d] for contiguous B loads
__global__ void k_cvt_w(const float* __restrict__ w1, const float* __restrict__ w2,
                        const float* __restrict__ w3, __bf16* __restrict__ wbT) {
  const int per = HH * DDIM * HDD;
  int i = blockIdx.x * blockDim.x + threadIdx.x;
  int st = gridDim.x * blockDim.x;
  for (; i < 3 * per; i += st) {
    int pr = i / per, j = i % per;
    int d = j % DDIM;
    int he = j / DDIM;
    int e = he % HDD;
    int h = he / HDD;
    const float* src = (pr == 0) ? w1 : ((pr == 1) ? w2 : w3);
    wbT[i] = (__bf16)src[((size_t)h * DDIM + d) * HDD + e];
  }
}

// ln2d over (S,D) per batch; writes fp32 + bf16
__global__ void k_ln_in(const float* __restrict__ img, float* __restrict__ xf,
                        __bf16* __restrict__ xb) {
  int b = blockIdx.x, tid = threadIdx.x;
  const float* src = img + (size_t)b * SD;
  float* xo = xf + (size_t)b * SD;
  __bf16* xo16 = xb + (size_t)b * SD;
  __shared__ float red[512];
  __shared__ float s_mean, s_inv;
  float s = 0.f, ss = 0.f;
  for (int i = tid; i < SD; i += 512) { float v = src[i]; s += v; ss += v * v; }
  red[tid] = s; __syncthreads();
  for (int st = 256; st > 0; st >>= 1) { if (tid < st) red[tid] += red[tid + st]; __syncthreads(); }
  if (!tid) s_mean = red[0] / (float)SD;
  __syncthreads();
  red[tid] = ss; __syncthreads();
  for (int st = 256; st > 0; st >>= 1) { if (tid < st) red[tid] += red[tid + st]; __syncthreads(); }
  if (!tid) { float var = red[0] / (float)SD - s_mean * s_mean; s_inv = rsqrtf(var + EPSF); }
  __syncthreads();
  float mean = s_mean, inv = s_inv;
  for (int i = tid; i < SD; i += 512) {
    float v = (src[i] - mean) * inv;
    xo[i] = v; xo16[i] = (__bf16)v;
  }
}

// q/k/v projections fused (A fragments shared across the 3 GEMMs)
__global__ void k_qkv(const __bf16* __restrict__ xb, const __bf16* __restrict__ wbT,
                      float* __restrict__ qkvf, float* __restrict__ stats) {
  int idx = blockIdx.x;
  int mt = idx % MT; idx /= MT;
  int h = idx % HH;
  int b = idx / HH;
  int wv = threadIdx.x >> 5;     // 0..3 : N-tile
  int m0 = mt * 16;
  int mval = SS - m0;
  const __bf16* A = xb + (size_t)b * SD + (size_t)m0 * DDIM;
  const size_t hs = (size_t)HDD * DDIM;           // per-(proj,h) block in wbT
  const __bf16* B0 = wbT + (size_t)h * hs + (size_t)(wv * 16) * DDIM;
  v8f c0 = {}, c1 = {}, c2 = {};
#pragma unroll 2
  for (int kc = 0; kc < DDIM / 32; ++kc) {
    v16bf a = load_a_bf(A + kc * 32, DDIM, mval);
    c0 = wmma_bf16(a, load_bT_bf(B0 + kc * 32, DDIM), c0);
    c1 = wmma_bf16(a, load_bT_bf(B0 + (size_t)HH * hs + kc * 32, DDIM), c1);
    c2 = wmma_bf16(a, load_bT_bf(B0 + (size_t)2 * HH * hs + kc * 32, DDIM), c2);
  }
  int lane = threadIdx.x & 31;
  int n = wv * 16 + (lane & 15);
  int mh = (lane >> 4) << 3;
  v8f cs[3] = {c0, c1, c2};
#pragma unroll
  for (int p = 0; p < 3; ++p) {
    float* dst = qkvf + (size_t)p * NTOK + (size_t)(b * HH + h) * SS * HDD;
    float ls = 0.f, lss = 0.f;
#pragma unroll
    for (int r = 0; r < 8; ++r) {
      int m = m0 + r + mh;
      float v = cs[p][r];
      if (m < SS) {
        dst[(size_t)m * HDD + n] = v;
        ls += v; lss += v * v;
      }
    }
    float* st = stats + ((size_t)p * BB * HH + (size_t)b * HH + h) * 2;
    atomicAdd(st + 0, ls);
    atomicAdd(st + 1, lss);
  }
}

__global__ void k_qkv_ln(float* __restrict__ qkvf, const float* __restrict__ stats,
                         __bf16* __restrict__ qkvb) {
  size_t i = (size_t)blockIdx.x * blockDim.x + threadIdx.x;
  if (i >= (size_t)3 * NTOK) return;
  size_t proj = i / NTOK;
  size_t rem = i % NTOK;
  size_t bh = rem / ((size_t)SS * HDD);
  const float* st = stats + (proj * BB * HH + bh) * 2;
  const float cnt = (float)(SS * HDD);
  float mean = st[0] / cnt;
  float var  = st[1] / cnt - mean * mean;
  float inv  = rsqrtf(var + EPSF);
  float v = (qkvf[i] - mean) * inv;
  qkvf[i] = v;
  qkvb[i] = (__bf16)v;
}

// fused attention: one WG per (b,h,16-row q tile); K tile staged via TDM
__global__ void k_attn(const __bf16* __restrict__ qkvb, float* __restrict__ og) {
  __shared__ __bf16 kls[SPAD * HDD];   // 28 KB
  __shared__ float  sc[16 * SPAD];     // 14 KB; reused in place for bf16 att
  int idx = blockIdx.x;
  int mt = idx % MT; idx /= MT;
  int h = idx % HH;
  int b = idx / HH;
  int m0 = mt * 16;
  int tid = threadIdx.x;
  size_t headoff = (size_t)(b * HH + h) * SS * HDD;
  const __bf16* qh = qkvb + headoff;
  const __bf16* kh = qkvb + (size_t)NTOK + headoff;
  const __bf16* vh = qkvb + (size_t)2 * NTOK + headoff;
  // zero padding rows [SS, SPAD)
  for (int i = tid + SS * HDD; i < SPAD * HDD; i += blockDim.x) kls[i] = (__bf16)0.0f;
  // TDM: one wave issues the 1-D tile load of the whole K head block
  if (tid < 32) {
    tdm_load_1d_bf16(kh, (unsigned)(size_t)(void*)kls, SS * HDD);
  }
  __syncthreads();
  int wv = tid >> 5;
  int lane = tid & 31;
  int nn = lane & 15;
  int mh = (lane >> 4) << 3;
  const float scale = rsqrtf((float)SS);
  v16bf a0 = load_a_bf(qh + (size_t)m0 * HDD, HDD, SS - m0);
  v16bf a1 = load_a_bf(qh + (size_t)m0 * HDD + 32, HDD, SS - m0);
  for (int kt = wv; kt < MT; kt += 4) {
    v8f c = {};
    c = wmma_bf16(a0, load_bT_bf(kls + (size_t)(kt * 16) * HDD, HDD), c);
    c = wmma_bf16(a1, load_bT_bf(kls + (size_t)(kt * 16) * HDD + 32, HDD), c);
#pragma unroll
    for (int r = 0; r < 8; ++r)
      sc[(r + mh) * SPAD + kt * 16 + nn] = c[r] * scale;
  }
  __syncthreads();
  __bf16* attb = (__bf16*)sc;          // att row i at byte offset i*SPAD*4, stride 2*SPAD elems
  if (tid < 16) {
    int gm = m0 + tid;
    float* row = sc + tid * SPAD;
    __bf16* arow = (__bf16*)row;
    if (gm < SS) {
      float mx = -3.4e38f;
      for (int cix = 0; cix < SS; ++cix) mx = fmaxf(mx, row[cix]);
      float sum = 0.f;
      for (int cix = 0; cix < SS; ++cix) { float e = __expf(row[cix] - mx); row[cix] = e; sum += e; }
      float rs = 1.f / sum;
      for (int cix = 0; cix < SS; ++cix) arow[cix] = (__bf16)(row[cix] * rs);  // in-place: 2c <= 4c
      for (int cix = SS; cix < SPAD; ++cix) arow[cix] = (__bf16)0.0f;
    } else {
      for (int cix = 0; cix < SPAD; ++cix) arow[cix] = (__bf16)0.0f;
    }
  }
  __syncthreads();
  v8f c = {};
#pragma unroll
  for (int kc = 0; kc < SPAD / 32 - 1; ++kc) {   // kc 0..5: fully in-range K
    v16bf a  = load_a_full(attb + kc * 32, 2 * SPAD);
    v16bf bf = load_b_bf(vh + (size_t)(kc * 32) * HDD + wv * 16, HDD);
    c = wmma_bf16(a, bf, c);
  }
  { // last chunk: clamp K rows (att columns there are zero)
    const int kc = SPAD / 32 - 1;
    v16bf a  = load_a_full(attb + kc * 32, 2 * SPAD);
    v16bf bf = load_b_clamp(vh + (size_t)(kc * 32) * HDD + wv * 16, HDD, SS - kc * 32);
    c = wmma_bf16(a, bf, c);
  }
#pragma unroll
  for (int r = 0; r < 8; ++r) {
    int m = m0 + r + mh;
    if (m < SS)
      og[((size_t)b * SS + m) * DDIM + h * HDD + wv * 16 + nn] = c[r];
  }
}

// residual + ln2d over (S,D); writes skip (fp32) and y (bf16)
__global__ void k_out_ln(const float* __restrict__ og, const float* __restrict__ xf,
                         float* __restrict__ skip, __bf16* __restrict__ yb) {
  int b = blockIdx.x, tid = threadIdx.x;
  const float* o = og + (size_t)b * SD;
  const float* x = xf + (size_t)b * SD;
  float* sk = skip + (size_t)b * SD;
  __bf16* y = yb + (size_t)b * SD;
  __shared__ float red[512];
  __shared__ float s_mean, s_inv;
  float s = 0.f, ss = 0.f;
  for (int i = tid; i < SD; i += 512) { float t = o[i] + x[i]; sk[i] = t; s += t; ss += t * t; }
  red[tid] = s; __syncthreads();
  for (int st = 256; st > 0; st >>= 1) { if (tid < st) red[tid] += red[tid + st]; __syncthreads(); }
  if (!tid) s_mean = red[0] / (float)SD;
  __syncthreads();
  red[tid] = ss; __syncthreads();
  for (int st = 256; st > 0; st >>= 1) { if (tid < st) red[tid] += red[tid + st]; __syncthreads(); }
  if (!tid) { float var = red[0] / (float)SD - s_mean * s_mean; s_inv = rsqrtf(var + EPSF); }
  __syncthreads();
  float mean = s_mean, inv = s_inv;
  for (int i = tid; i < SD; i += 512) y[i] = (__bf16)((sk[i] - mean) * inv);
}

// MLP1: [32 x 75648] @ W1[75648 x 1024], K split across grid, fp32 atomic partials
__global__ void k_mlp1(const __bf16* __restrict__ yb, const float* __restrict__ W1,
                       float* __restrict__ hacc) {
  int ng = blockIdx.x & 15;
  int ks = blockIdx.x >> 4;
  int wv = threadIdx.x >> 5;
  int n0 = (ng * 4 + wv) * 16;
  int kc0 = ks * KPER;
  int kc1 = kc0 + KPER; if (kc1 > KC1) kc1 = KC1;
  v8f c0 = {}, c1 = {};
  for (int kc = kc0; kc < kc1; ++kc) {
    if (kc + 1 < kc1) __builtin_prefetch(W1 + (size_t)(kc + 1) * 32 * HID + n0, 0, 0);
    v16bf bf = load_b_f32(W1 + (size_t)kc * 32 * HID + n0, HID);
    v16bf a0 = load_a_full(yb + (size_t)kc * 32, SD);
    v16bf a1 = load_a_full(yb + (size_t)16 * SD + (size_t)kc * 32, SD);
    c0 = wmma_bf16(a0, bf, c0);
    c1 = wmma_bf16(a1, bf, c1);
  }
  int lane = threadIdx.x & 31;
  int n = n0 + (lane & 15);
  int mh = (lane >> 4) << 3;
#pragma unroll
  for (int r = 0; r < 8; ++r) {
    atomicAdd(&hacc[(size_t)(r + mh) * HID + n], c0[r]);
    atomicAdd(&hacc[(size_t)(16 + r + mh) * HID + n], c1[r]);
  }
}

__global__ void k_gelu_h(const float* __restrict__ hacc, const float* __restrict__ b1,
                         __bf16* __restrict__ hb) {
  int i = blockIdx.x * blockDim.x + threadIdx.x;
  if (i >= BB * HID) return;
  float v = hacc[i] + b1[i % HID];
  hb[i] = (__bf16)gelu_exact(v);
}

// MLP2: [32 x 1024] @ W2[1024 x 75648] + bias, GELU, + skip -> out
__global__ void k_mlp2(const __bf16* __restrict__ hb, const float* __restrict__ W2,
                       const float* __restrict__ b2, const float* __restrict__ skip,
                       float* __restrict__ out) {
  int wv = threadIdx.x >> 5;
  int n0 = (blockIdx.x * 4 + wv) * 16;
  v8f c0 = {}, c1 = {};
  for (int kc = 0; kc < HID / 32; ++kc) {
    if (kc + 1 < HID / 32) __builtin_prefetch(W2 + (size_t)(kc + 1) * 32 * SD + n0, 0, 0);
    v16bf bf = load_b_f32(W2 + (size_t)kc * 32 * SD + n0, SD);
    v16bf a0 = load_a_full(hb + (size_t)kc * 32, HID);
    v16bf a1 = load_a_full(hb + (size_t)16 * HID + (size_t)kc * 32, HID);
    c0 = wmma_bf16(a0, bf, c0);
    c1 = wmma_bf16(a1, bf, c1);
  }
  int lane = threadIdx.x & 31;
  int col = n0 + (lane & 15);
  int mh = (lane >> 4) << 3;
  float bias = b2[col];
#pragma unroll
  for (int r = 0; r < 8; ++r) {
    int m = r + mh;
    float g0 = gelu_exact(c0[r] + bias);
    out[(size_t)m * SD + col] = skip[(size_t)m * SD + col] + g0;
    float g1 = gelu_exact(c1[r] + bias);
    out[(size_t)(16 + m) * SD + col] = skip[(size_t)(16 + m) * SD + col] + g1;
  }
}

extern "C" void kernel_launch(void* const* d_in, const int* in_sizes, int n_in,
                              void* d_out, int out_size, void* d_ws, size_t ws_size,
                              hipStream_t stream) {
  (void)in_sizes; (void)n_in; (void)out_size; (void)ws_size;
  const float* images = (const float*)d_in[0];
  const float* w1  = (const float*)d_in[1];
  const float* w2  = (const float*)d_in[2];
  const float* w3  = (const float*)d_in[3];
  const float* mw1 = (const float*)d_in[4];
  const float* mb1 = (const float*)d_in[5];
  const float* mw2 = (const float*)d_in[6];
  const float* mb2 = (const float*)d_in[7];
  float* out = (float*)d_out;

  char* base = (char*)d_ws;
  size_t off = 0;
  auto take = [&](size_t bytes) -> void* {
    void* r = (void*)(base + off);
    off += (bytes + 255) & ~(size_t)255;
    return r;
  };
  float*  xf    = (float*) take((size_t)NTOK * 4);
  __bf16* xb    = (__bf16*)take((size_t)NTOK * 2);
  __bf16* wbT   = (__bf16*)take((size_t)3 * HH * DDIM * HDD * 2);
  float*  qkvf  = (float*) take((size_t)3 * NTOK * 4);
  __bf16* qkvb  = (__bf16*)take((size_t)3 * NTOK * 2);
  float*  stats = (float*) take((size_t)3 * BB * HH * 2 * 4);
  float*  og    = (float*) take((size_t)NTOK * 4);
  float*  skip  = (float*) take((size_t)NTOK * 4);
  __bf16* yb    = (__bf16*)take((size_t)NTOK * 2);
  float*  hacc  = (float*) take((size_t)BB * HID * 4);
  __bf16* hb    = (__bf16*)take((size_t)BB * HID * 2);

  k_zero<<<dim3(8),   dim3(256), 0, stream>>>(stats, 3 * BB * HH * 2);
  k_zero<<<dim3(128), dim3(256), 0, stream>>>(hacc, BB * HID);
  k_ln_in<<<dim3(BB), dim3(512), 0, stream>>>(images, xf, xb);
  k_cvt_w<<<dim3(432), dim3(256), 0, stream>>>(w1, w2, w3, wbT);
  k_qkv<<<dim3(BB * HH * MT), dim3(128), 0, stream>>>(xb, wbT, qkvf, stats);
  {
    int total = 3 * NTOK;
    k_qkv_ln<<<dim3((total + 255) / 256), dim3(256), 0, stream>>>(qkvf, stats, qkvb);
  }
  k_attn<<<dim3(BB * HH * MT), dim3(128), 0, stream>>>(qkvb, og);
  k_out_ln<<<dim3(BB), dim3(512), 0, stream>>>(og, xf, skip, yb);
  k_mlp1<<<dim3(16 * KSPLIT), dim3(128), 0, stream>>>(yb, mw1, hacc);
  k_gelu_h<<<dim3((BB * HID + 255) / 256), dim3(256), 0, stream>>>(hacc, mb1, hb);
  k_mlp2<<<dim3(SD / 64), dim3(128), 0, stream>>>(hb, mw2, mb2, skip, out);
}
